// PeakSetSIGReg_20976620274239
// MI455X (gfx1250) — matrix operand order repaired and checked by reading, not compiled, not run on hardware
//
#include <hip/hip_runtime.h>
#include <hip/hip_fp16.h>

typedef __attribute__((ext_vector_type(16))) _Float16 v16h;
typedef __attribute__((ext_vector_type(8)))  _Float16 v8h;
typedef __attribute__((ext_vector_type(4)))  _Float16 v4h;
typedef __attribute__((ext_vector_type(8)))  float    v8f;

#define B_  16
#define N_  1024
#define H_  8
#define D_  64
#define QT  128          // query rows per workgroup (8 waves x 16 rows)
#define KT  64           // key tile per iteration == softmax chunk width
#define LDK 72           // padded LDS row stride in halves (144B, spreads banks)
#define NEGBIG (-1.0e30f)

// load halves p[0..7] and p[16..23] (two 16B LDS loads) into a v16h
__device__ __forceinline__ v16h ld_contig16(const _Float16* p) {
    v8h lo = *(const v8h*)(p);
    v8h hi = *(const v8h*)(p + 16);
    v16h r;
#pragma unroll
    for (int i = 0; i < 8; ++i) { r[i] = lo[i]; r[8 + i] = hi[i]; }
    return r;
}

// A-fragment (16x32 f16): lane halves [0..7] = K = d0..d0+7, [8..15] = K = d0+16..d0+23
__device__ __forceinline__ v16h ld_a_frag(const _Float16* rowp, int d0) {
    v8h lo = *(const v8h*)(rowp + d0);
    v8h hi = *(const v8h*)(rowp + d0 + 16);
    v16h r;
#pragma unroll
    for (int i = 0; i < 8; ++i) { r[i] = lo[i]; r[8 + i] = hi[i]; }
    return r;
}

// max-reduce across each 16-lane group via DPP only (no LDS / bpermute):
// xor1 (quad_perm 0xB1), xor2 (quad_perm 0x4E),
// row_half_mirror (0x141, pairs the two quads in each 8-group),
// row_mirror (0x140, pairs the two 8-groups in each row of 16)
#define DPP_F32(x, ctrl) \
    __builtin_bit_cast(float, __builtin_amdgcn_update_dpp( \
        __builtin_bit_cast(int, (x)), __builtin_bit_cast(int, (x)), (ctrl), 0xf, 0xf, true))
__device__ __forceinline__ float row16_max(float x) {
    x = fmaxf(x, DPP_F32(x, 0xB1));
    x = fmaxf(x, DPP_F32(x, 0x4E));
    x = fmaxf(x, DPP_F32(x, 0x141));
    x = fmaxf(x, DPP_F32(x, 0x140));
    return x;
}

__global__ __launch_bounds__(256) void fused_rope_flash_attn(
    const float* __restrict__ xq, const float* __restrict__ xk,
    const float* __restrict__ xv, const int* __restrict__ vis,
    float* __restrict__ out)
{
    __shared__ __align__(16) _Float16 Qs[QT][LDK];      // RoPE'd Q, row-major [row][d]
    __shared__ __align__(16) _Float16 Ks[KT][LDK];      // RoPE'd K, row-major [key][d]
    __shared__ __align__(16) _Float16 Vs[D_][LDK];      // V transposed [d][key]
    __shared__ __align__(16) _Float16 Ps[8][16][LDK];   // per-wave P strip [row][key64]
    __shared__ float kvisf[KT];                         // additive key mask (0 / -1e30)
    __shared__ float Lr[8][16];                         // per-wave row sums (epilogue)

    const int qblk = blockIdx.x;   // 0..7
    const int h    = blockIdx.y;   // 0..7
    const int b    = blockIdx.z;   // 0..15
    const int tid  = threadIdx.x;
    const int wave = tid >> 5;
    const int lane = tid & 31;
    const int lm   = lane & 15;
    const int hs   = lane >> 4;    // half-select (lane group)
    const int q0   = qblk * QT;

    const float FREQ = -0.28782313662425572f;  // -ln(10000)/32
    const float SM   = 0.125f;                 // 1/sqrt(64), folded into Q

    // float4 staging: quad index pq = i & 15 is loop-invariant (stride 256 ≡ 0 mod 16)
    const int   pq = tid & 15;                 // which float4 within a 64-float row
    const float f0 = __expf((float)(2 * pq) * FREQ);
    const float f1 = __expf((float)(2 * pq + 1) * FREQ);

    // ---- stage RoPE'd, pre-scaled Q tile (128x64 -> f16), b128 loads ----
#pragma unroll 4
    for (int i = tid; i < QT * (D_ / 4); i += 256) {
        int row = i >> 4;            // 0..127
        int n   = q0 + row;
        float4 x = *(const float4*)(xq + ((((size_t)b * N_ + n) * H_ + h) * D_ + 4 * pq));
        float sn0, cs0, sn1, cs1;
        __sincosf((float)n * f0, &sn0, &cs0);
        __sincosf((float)n * f1, &sn1, &cs1);
        v4h w;
        w[0] = (_Float16)((x.x * cs0 - x.y * sn0) * SM);
        w[1] = (_Float16)((x.y * cs0 + x.x * sn0) * SM);
        w[2] = (_Float16)((x.z * cs1 - x.w * sn1) * SM);
        w[3] = (_Float16)((x.w * cs1 + x.z * sn1) * SM);
        *(v4h*)&Qs[row][4 * pq] = w;
    }
    __syncthreads();

    // ---- per-wave Q A-fragments (rows wave*16 .. +15) ----
    const int qrow = wave * 16 + lm;         // A layout: lane -> row l%16
    v16h qa[2];
#pragma unroll
    for (int c = 0; c < 2; ++c)
        qa[c] = ld_a_frag(&Qs[qrow][0], c * 32 + hs * 8);

    // constant "ones-column" B fragment: B[k][0] = 1, else 0 -> row-sum via WMMA
    v16h vones;
#pragma unroll
    for (int t = 0; t < 16; ++t) vones[t] = (lm == 0) ? (_Float16)1.0f : (_Float16)0.0f;

    // ---- flash-attention running state (C layout: VGPR r -> row r + hs*8) ----
    float mrow[8];
    v8f o[4], o4;   // o4 accumulates row sums l via ones-column WMMA
#pragma unroll
    for (int r = 0; r < 8; ++r) { mrow[r] = NEGBIG; o4[r] = 0.0f; }
#pragma unroll
    for (int f = 0; f < 4; ++f)
#pragma unroll
        for (int r = 0; r < 8; ++r) o[f][r] = 0.0f;

    // ---- main loop: one 64-key tile == one softmax update ----
#pragma unroll 1
    for (int kt = 0; kt < N_; kt += KT) {
        __syncthreads();   // previous tile's LDS reads complete before refill

        // stage RoPE'd K tile (row-major), b128 loads / b64 LDS stores
#pragma unroll 2
        for (int i = tid; i < KT * (D_ / 4); i += 256) {
            int row = i >> 4;
            int n   = kt + row;
            float4 x = *(const float4*)(xk + ((((size_t)b * N_ + n) * H_ + h) * D_ + 4 * pq));
            float sn0, cs0, sn1, cs1;
            __sincosf((float)n * f0, &sn0, &cs0);
            __sincosf((float)n * f1, &sn1, &cs1);
            v4h w;
            w[0] = (_Float16)(x.x * cs0 - x.y * sn0);
            w[1] = (_Float16)(x.y * cs0 + x.x * sn0);
            w[2] = (_Float16)(x.z * cs1 - x.w * sn1);
            w[3] = (_Float16)(x.w * cs1 + x.z * sn1);
            *(v4h*)&Ks[row][4 * pq] = w;
        }
        // stage V tile transposed: Vs[d][key]
#pragma unroll 2
        for (int i = tid; i < KT * (D_ / 4); i += 256) {
            int row = i >> 4;
            float4 x = *(const float4*)(xv + ((((size_t)b * N_ + (kt + row)) * H_ + h) * D_ + 4 * pq));
            Vs[4 * pq + 0][row] = (_Float16)x.x;
            Vs[4 * pq + 1][row] = (_Float16)x.y;
            Vs[4 * pq + 2][row] = (_Float16)x.z;
            Vs[4 * pq + 3][row] = (_Float16)x.w;
        }
        if (tid < KT)
            kvisf[tid] = (vis[b * N_ + kt + tid] != 0) ? 0.0f : NEGBIG;
        __syncthreads();

        // ---- S = Q K^T : 16 rows x 64 keys (four 16x16 C tiles) ----
        v8f s[4];
#pragma unroll
        for (int t = 0; t < 4; ++t)
#pragma unroll
            for (int r = 0; r < 8; ++r) s[t][r] = 0.0f;
#pragma unroll
        for (int c = 0; c < 2; ++c) {
#pragma unroll
            for (int t = 0; t < 4; ++t) {
                // B layout (32x16): lane col = key t*16 + l%16; halves -> K = hs*8 + idx, 16 + hs*8 + idx
                v16h bk = ld_contig16(&Ks[t * 16 + lm][c * 32 + hs * 8]);
                s[t] = __builtin_amdgcn_wmma_f32_16x16x32_f16(
                        false, qa[c], false, bk, (short)0, s[t], false, false);
            }
        }

        // ---- one online-softmax update for all 64 keys ----
        float kv[4];
#pragma unroll
        for (int t = 0; t < 4; ++t) kv[t] = kvisf[t * 16 + lm];
        float alpha[8];
#pragma unroll
        for (int r = 0; r < 8; ++r) {
            float v0 = s[0][r] + kv[0];
            float v1 = s[1][r] + kv[1];
            float v2 = s[2][r] + kv[2];
            float v3 = s[3][r] + kv[3];
            float mx   = row16_max(fmaxf(fmaxf(v0, v1), fmaxf(v2, v3)));
            float mnew = fmaxf(mrow[r], mx);
            alpha[r]   = __expf(mrow[r] - mnew);
            mrow[r]    = mnew;
            int prow = r + hs * 8;                  // C-layout row
            Ps[wave][prow][lm]      = (_Float16)__expf(v0 - mnew);
            Ps[wave][prow][16 + lm] = (_Float16)__expf(v1 - mnew);
            Ps[wave][prow][32 + lm] = (_Float16)__expf(v2 - mnew);
            Ps[wave][prow][48 + lm] = (_Float16)__expf(v3 - mnew);
        }
#pragma unroll
        for (int r = 0; r < 8; ++r) {
            o4[r] *= alpha[r];
#pragma unroll
            for (int f = 0; f < 4; ++f) o[f][r] *= alpha[r];
        }

        // ---- O += P V ; l += P * ones  (A = P 16x64 via LDS transpose) ----
        v16h pa0 = ld_a_frag(&Ps[wave][lm][0], hs * 8);
        v16h pa1 = ld_a_frag(&Ps[wave][lm][0], 32 + hs * 8);
#pragma unroll
        for (int f = 0; f < 4; ++f) {
            v16h vb0 = ld_contig16(&Vs[f * 16 + lm][hs * 8]);
            v16h vb1 = ld_contig16(&Vs[f * 16 + lm][32 + hs * 8]);
            o[f] = __builtin_amdgcn_wmma_f32_16x16x32_f16(
                    false, pa0, false, vb0, (short)0, o[f], false, false);
            o[f] = __builtin_amdgcn_wmma_f32_16x16x32_f16(
                    false, pa1, false, vb1, (short)0, o[f], false, false);
        }
        o4 = __builtin_amdgcn_wmma_f32_16x16x32_f16(
                false, pa0, false, vones, (short)0, o4, false, false);
        o4 = __builtin_amdgcn_wmma_f32_16x16x32_f16(
                false, pa1, false, vones, (short)0, o4, false, false);
    }

    // ---- epilogue: broadcast row sums, normalize, zero invisible q rows ----
    if (lm == 0) {
#pragma unroll
        for (int r = 0; r < 8; ++r) Lr[wave][r + hs * 8] = o4[r];
    }
    // wave-private LDS strip; LDS ops are in-order within a wave
#pragma unroll
    for (int r = 0; r < 8; ++r) {
        int row = wave * 16 + r + hs * 8;   // local q row (C layout)
        int n   = q0 + row;
        float l   = Lr[wave][r + hs * 8];
        float inv = (l > 0.0f) ? (1.0f / l) : 0.0f;
        float qv  = (vis[b * N_ + n] != 0) ? 1.0f : 0.0f;
        float sc  = inv * qv;
        float* dst = out + (((size_t)b * N_ + n) * H_ + h) * D_;
#pragma unroll
        for (int f = 0; f < 4; ++f)
            dst[f * 16 + lm] = o[f][r] * sc;
    }
}

extern "C" void kernel_launch(void* const* d_in, const int* in_sizes, int n_in,
                              void* d_out, int out_size, void* d_ws, size_t ws_size,
                              hipStream_t stream) {
    const float* xq  = (const float*)d_in[0];
    const float* xk  = (const float*)d_in[1];
    const float* xv  = (const float*)d_in[2];
    const int*   vis = (const int*)d_in[3];
    float* out = (float*)d_out;
    (void)in_sizes; (void)n_in; (void)out_size; (void)d_ws; (void)ws_size;

    dim3 grid(N_ / QT, H_, B_);   // (8, 8, 16)
    dim3 block(256);
    fused_rope_flash_attn<<<grid, block, 0, stream>>>(xq, xk, xv, vis, out);
}
